// CNNDiscriminator_10445360464110
// MI455X (gfx1250) — matrix operand order repaired and checked by reading, not compiled, not run on hardware
//
#include <hip/hip_runtime.h>
#include <math.h>

// ---------------- problem constants (from reference) ----------------
constexpr int B       = 1024;
constexpr int S       = 128;
constexpr int E       = 256;
constexpr int F       = 512;     // filters per branch
constexpr int NFEAT   = 1536;    // 3 * 512
constexpr int NINTER  = 1024;
// branch K dims: h*E -> 768, 1024, 1280 ; T = S-h+1 -> 126, 125, 124

// ---------------- types ----------------
typedef unsigned short u16;
typedef __attribute__((ext_vector_type(8)))  u16   v8u16;
typedef __attribute__((ext_vector_type(16))) u16   v16u16;
typedef __attribute__((ext_vector_type(16))) __bf16 v16bf;
typedef __attribute__((ext_vector_type(8)))  float v8f;

__device__ __forceinline__ u16 f32_to_bf16(float f) {
  unsigned u = __builtin_bit_cast(unsigned, f);
  unsigned r = u + 0x7FFFu + ((u >> 16) & 1u);   // round-to-nearest-even
  return (u16)(r >> 16);
}

__device__ __forceinline__ v16bf pack16(v8u16 lo, v8u16 hi) {
  v16u16 t;
#pragma unroll
  for (int i = 0; i < 8; ++i) { t[i] = lo[i]; t[i + 8] = hi[i]; }
  return __builtin_bit_cast(v16bf, t);
}

// A fragment (16x32 bf16, MxK): lane<16 -> row M=lane, K = k0+{0..15}
//                               lane>=16 -> row M=lane-16, K = k0+{16..31}
__device__ __forceinline__ v16bf load_afrag(const u16* __restrict__ row, int k0, int half) {
  v8u16 lo = *(const v8u16*)(row + k0 + half * 16);
  v8u16 hi = *(const v8u16*)(row + k0 + half * 16 + 8);
  return pack16(lo, hi);
}

// B fragment (32x16 bf16, KxN): lane holds column N=lane&15, 16 contiguous K
// starting at k0 + (lane>>4)*16; W stored row-major [N][K] so this is one
// contiguous 32-byte read per lane.
__device__ __forceinline__ v16bf load_bfrag(const u16* __restrict__ p) {
  v8u16 lo = *(const v8u16*)(p);
  v8u16 hi = *(const v8u16*)(p + 8);
  return pack16(lo, hi);
}

// ---------------- kernel 1: embedding gather + f32->bf16 ----------------
__global__ void k_embed(const int* __restrict__ x, const float* __restrict__ emb,
                        u16* __restrict__ Xe) {
  const int row = blockIdx.x;             // 0..B*S-1
  const int e   = threadIdx.x;            // 0..255 (== E)
  const int tok = x[row];
  const float v = emb[(size_t)tok * E + e];
  Xe[(size_t)row * E + e] = f32_to_bf16(v);
}

// ---------------- kernel 2: generic f32 -> bf16 convert ----------------
__global__ void k_cvt_bf16(const float* __restrict__ src, u16* __restrict__ dst, int n) {
  int i = blockIdx.x * 256 + threadIdx.x;
  if (i < n) dst[i] = f32_to_bf16(src[i]);
}

// ---------------- kernel 3: conv branch as implicit-im2col WMMA GEMM ------
// Workgroup = one batch element b; wave w owns filters [w*64, w*64+64).
// Loops t in chunks of 16 rows (M), K in steps of 32 with bf16 WMMA.
// Max-pool fused via running per-column max of raw accumulators
// (tanh is monotonic: max_t tanh(y+b) == tanh(max_t y + b)).
template <int KDIM, int TDIM>
__global__ __launch_bounds__(256) void k_conv(const u16* __restrict__ Xe,
                                              const u16* __restrict__ W,
                                              const float* __restrict__ bias,
                                              u16* __restrict__ feats,
                                              int branch_off) {
  const int b    = blockIdx.x;
  const int tid  = threadIdx.x;
  const int wave = tid >> 5;
  const int lane = tid & 31;
  const int half = lane >> 4;   // which K half this lane carries
  const int mrow = lane & 15;   // A-row within tile / output column within tile

  int   ncol[4];
  float bcol[4];
#pragma unroll
  for (int j = 0; j < 4; ++j) {
    ncol[j] = wave * 64 + j * 16 + mrow;
    bcol[j] = bias[ncol[j]];
  }

  float maxv[4] = {-INFINITY, -INFINITY, -INFINITY, -INFINITY};

  const int nchunks = (TDIM + 15) / 16;
  for (int c = 0; c < nchunks; ++c) {
    int t = c * 16 + mrow;
    if (t > TDIM - 1) t = TDIM - 1;            // tail rows duplicate last t; max-safe
    const u16* arow = Xe + ((size_t)b * S + t) * E;   // contiguous im2col row (h*E)

    v8f acc[4] = {};
    for (int k0 = 0; k0 < KDIM; k0 += 32) {
      v16bf a = load_afrag(arow, k0, half);
      if (k0 + 32 < KDIM) __builtin_prefetch(arow + k0 + 32, 0, 1);
#pragma unroll
      for (int j = 0; j < 4; ++j) {
        v16bf bb = load_bfrag(W + (size_t)ncol[j] * KDIM + k0 + half * 16);
        acc[j] = __builtin_amdgcn_wmma_f32_16x16x32_bf16(
            false, a, false, bb, (short)0, acc[j], false, false);
      }
    }
#pragma unroll
    for (int j = 0; j < 4; ++j)
#pragma unroll
      for (int r = 0; r < 8; ++r) maxv[j] = fmaxf(maxv[j], acc[j][r]);
  }

  // combine the two half-wave copies of each column (M rows 0..7 vs 8..15)
#pragma unroll
  for (int j = 0; j < 4; ++j) {
    float o = __shfl_xor(maxv[j], 16, 32);
    maxv[j] = fmaxf(maxv[j], o);
  }

  if (half == 0) {
#pragma unroll
    for (int j = 0; j < 4; ++j) {
      float feat = tanhf(maxv[j] + bcol[j]);
      feats[(size_t)b * NFEAT + branch_off + ncol[j]] = f32_to_bf16(feat);
    }
  }
}

// ---------------- kernel 4: FC1 (1536 -> 1024) + sigmoid, WMMA ------------
// grid = (B/16, NINTER/512); workgroup = 16 batch rows x 512 output cols.
__global__ __launch_bounds__(256) void k_fc1(const u16* __restrict__ feats,
                                             const u16* __restrict__ Wfc1,
                                             const float* __restrict__ bias,
                                             float* __restrict__ hbuf) {
  const int mbase = blockIdx.x * 16;
  const int nbase = blockIdx.y * 512;
  const int tid   = threadIdx.x;
  const int wave  = tid >> 5;
  const int lane  = tid & 31;
  const int half  = lane >> 4;
  const int mrow  = lane & 15;

  const u16* arow = feats + (size_t)(mbase + mrow) * NFEAT;

  int   ncol[4];
  float bcol[4];
#pragma unroll
  for (int j = 0; j < 4; ++j) {
    ncol[j] = nbase + wave * 64 + j * 16 + mrow;
    bcol[j] = bias[ncol[j]];
  }

  v8f acc[4] = {};
  for (int k0 = 0; k0 < NFEAT; k0 += 32) {
    v16bf a = load_afrag(arow, k0, half);
#pragma unroll
    for (int j = 0; j < 4; ++j) {
      v16bf bb = load_bfrag(Wfc1 + (size_t)ncol[j] * NFEAT + k0 + half * 16);
      acc[j] = __builtin_amdgcn_wmma_f32_16x16x32_bf16(
          false, a, false, bb, (short)0, acc[j], false, false);
    }
  }

  // D layout: VGPR r, lane -> (m = mbase + half*8 + r, n = ncol[j])
#pragma unroll
  for (int j = 0; j < 4; ++j) {
#pragma unroll
    for (int r = 0; r < 8; ++r) {
      float v = acc[j][r] + bcol[j];
      v = 1.0f / (1.0f + __expf(-v));
      hbuf[(size_t)(mbase + half * 8 + r) * NINTER + ncol[j]] = v;
    }
  }
}

// ---------------- kernel 5: FC2 (1024 -> 2) + softmax --------------------
__global__ void k_fc2(const float* __restrict__ hbuf, const float* __restrict__ w2,
                      const float* __restrict__ b2, float* __restrict__ out) {
  const int b = blockIdx.x * blockDim.x + threadIdx.x;   // 0..B-1
  if (b >= B) return;
  const float* hr = hbuf + (size_t)b * NINTER;
  float a0 = b2[0], a1 = b2[1];
  for (int k = 0; k < NINTER; ++k) {
    const float h = hr[k];
    a0 += h * w2[k];
    a1 += h * w2[NINTER + k];
  }
  const float m  = fmaxf(a0, a1);
  const float e0 = __expf(a0 - m), e1 = __expf(a1 - m);
  const float s  = e0 + e1;
  out[b * 2 + 0] = e0 / s;
  out[b * 2 + 1] = e1 / s;
}

// ---------------- host launch --------------------------------------------
extern "C" void kernel_launch(void* const* d_in, const int* in_sizes, int n_in,
                              void* d_out, int out_size, void* d_ws, size_t ws_size,
                              hipStream_t stream) {
  const int*   x       = (const int*)  d_in[0];
  const float* emb     = (const float*)d_in[1];
  const float* w_conv0 = (const float*)d_in[2];
  const float* b_conv0 = (const float*)d_in[3];
  const float* w_conv1 = (const float*)d_in[4];
  const float* b_conv1 = (const float*)d_in[5];
  const float* w_conv2 = (const float*)d_in[6];
  const float* b_conv2 = (const float*)d_in[7];
  const float* w_fc1   = (const float*)d_in[8];
  const float* b_fc1   = (const float*)d_in[9];
  const float* w_fc2   = (const float*)d_in[10];
  const float* b_fc2   = (const float*)d_in[11];
  float* out = (float*)d_out;

  // workspace layout (bf16 stored as u16); all offsets even -> 4B aligned
  u16* Xe    = (u16*)d_ws;                           // B*S*E       = 33554432
  u16* Wc0   = Xe   + (size_t)B * S * E;             // 512*768     =   393216
  u16* Wc1   = Wc0  + (size_t)F * 768;               // 512*1024    =   524288
  u16* Wc2   = Wc1  + (size_t)F * 1024;              // 512*1280    =   655360
  u16* Wfc1  = Wc2  + (size_t)F * 1280;              // 1024*1536   =  1572864
  u16* feats = Wfc1 + (size_t)NINTER * NFEAT;        // B*1536      =  1572864
  float* hbuf = (float*)(feats + (size_t)B * NFEAT); // B*1024 f32  ~ total 81 MB

  // 1) embedding gather + bf16 convert
  k_embed<<<B * S, 256, 0, stream>>>(x, emb, Xe);

  // 2) weight conversions to bf16 (layouts already [out][K] row-major)
  auto cvt = [&](const float* s, u16* d, int n) {
    k_cvt_bf16<<<(n + 255) / 256, 256, 0, stream>>>(s, d, n);
  };
  cvt(w_conv0, Wc0, F * 768);
  cvt(w_conv1, Wc1, F * 1024);
  cvt(w_conv2, Wc2, F * 1280);
  cvt(w_fc1, Wfc1, NINTER * NFEAT);

  // 3) three conv branches (implicit im2col WMMA GEMM + fused tanh/maxpool)
  k_conv< 768, 126><<<B, 256, 0, stream>>>(Xe, Wc0, b_conv0, feats, 0);
  k_conv<1024, 125><<<B, 256, 0, stream>>>(Xe, Wc1, b_conv1, feats, 512);
  k_conv<1280, 124><<<B, 256, 0, stream>>>(Xe, Wc2, b_conv2, feats, 1024);

  // 4) FC1 + sigmoid
  k_fc1<<<dim3(B / 16, NINTER / 512), 256, 0, stream>>>(feats, Wfc1, b_fc1, hbuf);

  // 5) FC2 + softmax
  k_fc2<<<(B + 255) / 256, 256, 0, stream>>>(hbuf, w_fc2, b_fc2, out);
}
